// VQVAEBlock_32564442038320
// MI455X (gfx1250) — matrix operand order repaired and checked by reading, not compiled, not run on hardware
//
#include <hip/hip_runtime.h>
#include <hip/hip_bf16.h>
#include <stdint.h>

typedef __attribute__((ext_vector_type(16))) __bf16 v16bf;
typedef __attribute__((ext_vector_type(8)))  float  v8f;

#define D_DIM   512
#define N_TOK   8192
#define N_CODE  8192
#define KBLK    16          // 512 / 32 k-steps per code tile
#define NTILE   256         // 8192 codes / 32 per tile
#define LDSTR   520         // padded LDS row stride (halves)
#define OUT_Q_ELEMS 4194304 // 8*1024*512

static __device__ __forceinline__ __bf16 f2bf(float f) { return (__bf16)f; }

static __device__ __forceinline__ v16bf ldB(const __bf16* bt, int row, int kb, int koff) {
    union { uint4 u[2]; v16bf v; } bv;
    const uint4* p = reinterpret_cast<const uint4*>(bt + row * LDSTR + kb * 32 + koff);
    bv.u[0] = p[0];
    bv.u[1] = p[2];
    return bv.v;
}

// ---------------------------------------------------------------------------
// Kernel 1: w2[k] = sum_d codebook[k,d]^2   (one wave per code row)
// ---------------------------------------------------------------------------
__global__ __launch_bounds__(256) void vq_w2_kernel(const float* __restrict__ cb,
                                                    float* __restrict__ w2) {
    const int lane = threadIdx.x & 31;
    const int code = blockIdx.x * 8 + (threadIdx.x >> 5);
    const float4* p = reinterpret_cast<const float4*>(cb + code * D_DIM + lane * 16);
    float s = 0.f;
#pragma unroll
    for (int j = 0; j < 4; ++j) {
        float4 v = p[j];
        s += v.x * v.x + v.y * v.y + v.z * v.z + v.w * v.w;
    }
#pragma unroll
    for (int m = 1; m < 32; m <<= 1) s += __shfl_xor(s, m, 32);
    if (lane == 0) w2[code] = s;
}

// ---------------------------------------------------------------------------
// Kernel 2: argmin over codes via BF16 WMMA (dual-N, pipelined B, ping-pong
// LDS, prefetch), then gather + loss partials.
// Block = 128 threads (4 waves); wave w owns tokens [blk*64 + w*16, +16).
// ---------------------------------------------------------------------------
__global__ __launch_bounds__(128) void vq_main_kernel(const float* __restrict__ x,
                                                      const float* __restrict__ cb,
                                                      const float* __restrict__ w2,
                                                      float* __restrict__ out,
                                                      float* __restrict__ partials) {
    __shared__ __bf16 btile[2][32 * LDSTR];

    const int tid  = threadIdx.x;
    const int lane = tid & 31;
    const int wave = tid >> 5;
    const int t0   = blockIdx.x * 64 + wave * 16;

    // ---- Preload A fragments (ISA dense 16-bit A layout) -------------------
    const int arow = t0 + (lane & 15);
    const int aoff = (lane < 16) ? 0 : 8;
    v16bf afrag[KBLK];
#pragma unroll
    for (int kb = 0; kb < KBLK; ++kb) {
        const float* xp = x + arow * D_DIM + kb * 32 + aoff;
        float4 f0 = *reinterpret_cast<const float4*>(xp + 0);
        float4 f1 = *reinterpret_cast<const float4*>(xp + 4);
        float4 f2 = *reinterpret_cast<const float4*>(xp + 16);
        float4 f3 = *reinterpret_cast<const float4*>(xp + 20);
        v16bf a;
        a[0]  = f2bf(f0.x); a[1]  = f2bf(f0.y); a[2]  = f2bf(f0.z); a[3]  = f2bf(f0.w);
        a[4]  = f2bf(f1.x); a[5]  = f2bf(f1.y); a[6]  = f2bf(f1.z); a[7]  = f2bf(f1.w);
        a[8]  = f2bf(f2.x); a[9]  = f2bf(f2.y); a[10] = f2bf(f2.z); a[11] = f2bf(f2.w);
        a[12] = f2bf(f3.x); a[13] = f2bf(f3.y); a[14] = f2bf(f3.z); a[15] = f2bf(f3.w);
        afrag[kb] = a;
    }

    float minv[8];
    int   mini[8];
#pragma unroll
    for (int r = 0; r < 8; ++r) { minv[r] = 3.402823466e+38f; mini[r] = 0; }

    const float4* cb4 = reinterpret_cast<const float4*>(cb);
    const int crow = lane & 15;            // B: column N = lane&15
    const int koff = (lane < 16) ? 0 : 8;  // B: K 8-chunks at koff and koff+16 per lane-half

    // ---- Stage tile 0 ------------------------------------------------------
#pragma unroll 4
    for (int j = 0; j < 32; ++j) {
        int q  = j * 128 + tid;
        int r  = q >> 7;
        int dq = q & 127;
        float4 v = cb4[r * (D_DIM / 4) + dq];
        union { __bf16 h[4]; unsigned long long u; } pk;
        pk.h[0] = f2bf(v.x); pk.h[1] = f2bf(v.y);
        pk.h[2] = f2bf(v.z); pk.h[3] = f2bf(v.w);
        *reinterpret_cast<unsigned long long*>(&btile[0][r * LDSTR + dq * 4]) = pk.u;
    }
    __syncthreads();

    for (int tile = 0; tile < NTILE; ++tile) {
        const int cur   = tile & 1;
        const int nbase = tile * 32;
        const __bf16* bt = &btile[cur][0];

        // ---- Prefetch next tile's contiguous 64KB codebook region ----------
        if (tile + 1 < NTILE) {
            const char* nb = reinterpret_cast<const char*>(cb) +
                             (size_t)(tile + 1) * 32 * D_DIM * sizeof(float);
#pragma unroll
            for (int j = 0; j < 4; ++j)
                __builtin_prefetch(nb + (tid * 4 + j) * 128, 0, 3);
        }

        // ---- Dual 16x16 score tiles: pipelined B, 2 independent chains -----
        v8f acc0 = {0.f, 0.f, 0.f, 0.f, 0.f, 0.f, 0.f, 0.f};
        v8f acc1 = {0.f, 0.f, 0.f, 0.f, 0.f, 0.f, 0.f, 0.f};
        v16bf b0 = ldB(bt, crow, 0, koff);
        v16bf b1 = ldB(bt, crow + 16, 0, koff);
#pragma unroll
        for (int kb = 0; kb < KBLK; ++kb) {
            v16bf n0, n1;
            if (kb + 1 < KBLK) {
                n0 = ldB(bt, crow, kb + 1, koff);
                n1 = ldB(bt, crow + 16, kb + 1, koff);
            }
            acc0 = __builtin_amdgcn_wmma_f32_16x16x32_bf16(
                false, afrag[kb], false, b0, (short)0, acc0, false, false);
            acc1 = __builtin_amdgcn_wmma_f32_16x16x32_bf16(
                false, afrag[kb], false, b1, (short)0, acc1, false, false);
            if (kb + 1 < KBLK) { b0 = n0; b1 = n1; }
        }

        // ---- dist = w2[c] - 2*(x.w); running argmin per row-slot -----------
        const int   ci0 = nbase + crow;
        const int   ci1 = nbase + 16 + crow;
        const float w20 = w2[ci0];
        const float w21 = w2[ci1];
#pragma unroll
        for (int r = 0; r < 8; ++r) {
            float dv0 = w20 - 2.0f * acc0[r];
            if ((dv0 < minv[r]) || (dv0 == minv[r] && ci0 < mini[r])) {
                minv[r] = dv0; mini[r] = ci0;
            }
            float dv1 = w21 - 2.0f * acc1[r];
            if ((dv1 < minv[r]) || (dv1 == minv[r] && ci1 < mini[r])) {
                minv[r] = dv1; mini[r] = ci1;
            }
        }

        // ---- Stage next tile into the other LDS buffer ---------------------
        if (tile + 1 < NTILE) {
            const int nb2 = (tile + 1) * 32;
            __bf16* dst = &btile[cur ^ 1][0];
#pragma unroll 4
            for (int j = 0; j < 32; ++j) {
                int q  = j * 128 + tid;
                int r  = q >> 7;
                int dq = q & 127;
                float4 v = cb4[(nb2 + r) * (D_DIM / 4) + dq];
                union { __bf16 h[4]; unsigned long long u; } pk;
                pk.h[0] = f2bf(v.x); pk.h[1] = f2bf(v.y);
                pk.h[2] = f2bf(v.z); pk.h[3] = f2bf(v.w);
                *reinterpret_cast<unsigned long long*>(&dst[r * LDSTR + dq * 4]) = pk.u;
            }
        }
        __syncthreads();
    }

    // ---- Butterfly min-reduce across the 16 lanes sharing each row ---------
#pragma unroll
    for (int m = 1; m < 16; m <<= 1) {
#pragma unroll
        for (int r = 0; r < 8; ++r) {
            float ov = __shfl_xor(minv[r], m, 32);
            int   oi = __shfl_xor(mini[r], m, 32);
            if (ov < minv[r] || (ov == minv[r] && oi < mini[r])) {
                minv[r] = ov; mini[r] = oi;
            }
        }
    }

    // ---- Gather codebook[best] -> out, accumulate squared error ------------
    float lsum = 0.f;
    for (int m = 0; m < 16; ++m) {
        int best = __shfl(mini[m & 7], (m < 8) ? 0 : 16, 32);
        int tok  = t0 + m;
        const float4* qr = reinterpret_cast<const float4*>(cb + best * D_DIM);
        const float4* xr = reinterpret_cast<const float4*>(x + tok * D_DIM);
        float4*       orow = reinterpret_cast<float4*>(out + tok * D_DIM);
#pragma unroll 2
        for (int j = 0; j < 4; ++j) {
            int idx4 = lane * 4 + j * 32;   // 32 lanes * 4 float4 = 512 floats
            float4 q  = qr[idx4];
            float4 xv = xr[idx4];
            orow[idx4] = q;
            float dx = q.x - xv.x, dy = q.y - xv.y, dz = q.z - xv.z, dw = q.w - xv.w;
            lsum += dx * dx + dy * dy + dz * dz + dw * dw;
        }
    }
#pragma unroll
    for (int m = 1; m < 32; m <<= 1) lsum += __shfl_xor(lsum, m, 32);
    if (lane == 0) partials[blockIdx.x * 4 + wave] = lsum;
}

// ---------------------------------------------------------------------------
// Kernel 3: reduce 512 partials -> vqloss = 1.25 * sum / N
// ---------------------------------------------------------------------------
__global__ __launch_bounds__(256) void vq_loss_kernel(const float* __restrict__ partials,
                                                      float* __restrict__ loss_out) {
    __shared__ float s[256];
    int t = threadIdx.x;
    s[t] = partials[t] + partials[t + 256];
    __syncthreads();
    for (int st = 128; st > 0; st >>= 1) {
        if (t < st) s[t] += s[t + st];
        __syncthreads();
    }
    if (t == 0) loss_out[0] = s[0] * (1.25f / (float)OUT_Q_ELEMS);
}

extern "C" void kernel_launch(void* const* d_in, const int* in_sizes, int n_in,
                              void* d_out, int out_size, void* d_ws, size_t ws_size,
                              hipStream_t stream) {
    const float* x  = (const float*)d_in[0];  // [8,1024,512] f32
    const float* cb = (const float*)d_in[1];  // [8192,512]   f32
    float* out = (float*)d_out;               // 4194304 quant + 1 loss
    float* ws  = (float*)d_ws;
    float* w2       = ws;                     // 8192 floats
    float* partials = ws + N_CODE;            // 512 floats

    vq_w2_kernel<<<N_CODE / 8, 256, 0, stream>>>(cb, w2);
    vq_main_kernel<<<N_TOK / 64, 128, 0, stream>>>(x, cb, w2, out, partials);
    vq_loss_kernel<<<1, 256, 0, stream>>>(partials, out + OUT_Q_ELEMS);
}
